// ATConv_70042326663376
// MI455X (gfx1250) — matrix operand order repaired and checked by reading, not compiled, not run on hardware
//
#include <hip/hip_runtime.h>
#include <hip/hip_bf16.h>

#define B_   16
#define C_   256
#define H_   96
#define W_   96
#define HW_  (H_ * W_)       // 9216
#define K2_  9
#define BIN_ (HW_ / K2_)     // 1024

typedef __bf16 bf16;
typedef __attribute__((ext_vector_type(16))) __bf16 v16bf;
typedef __attribute__((ext_vector_type(8)))  __bf16 v8bf;
typedef __attribute__((ext_vector_type(8)))  float  v8f;

// ---------------------------------------------------------------------------
// 1) Average-pool x over 9 spatial bins of 1024 (feeds kernel synthesis).
//    pool(conv1x1(x)) == conv1x1(pool(x)): pool first, shrink the kp GEMM
//    from 256x256x9216 to 256x256x9 per batch.
// ---------------------------------------------------------------------------
__global__ void pool_kernel(const float* __restrict__ x, float* __restrict__ pooled) {
  // grid (K2_, C_, B_), block 256
  const int k = blockIdx.x, c = blockIdx.y, b = blockIdx.z;
  const float* p = x + ((size_t)(b * C_ + c)) * HW_ + (size_t)k * BIN_;
  float s = 0.f;
  for (int i = threadIdx.x; i < BIN_; i += 256) s += p[i];
  __shared__ float red[256];
  red[threadIdx.x] = s;
  __syncthreads();
  for (int off = 128; off > 0; off >>= 1) {
    if ((int)threadIdx.x < off) red[threadIdx.x] += red[threadIdx.x + off];
    __syncthreads();
  }
  if (threadIdx.x == 0) pooled[(b * C_ + c) * K2_ + k] = red[0] * (1.0f / BIN_);
}

// ---------------------------------------------------------------------------
// 2) Dynamic-kernel synthesis (tiny): kp-proj on pooled values, gelu,
//    9x9 linear, lateral inhibition. One block per batch, thread = channel.
// ---------------------------------------------------------------------------
__global__ void ker_synth(const float* __restrict__ pooled,
                          const float* __restrict__ w_kp, const float* __restrict__ b_kp,
                          const float* __restrict__ w_kg, const float* __restrict__ b_kg,
                          const float* __restrict__ diff_ctrl,
                          float* __restrict__ ker) {
  const int b = blockIdx.x;
  const int o = threadIdx.x;              // 256 threads == C_
  __shared__ float lp[C_ * K2_];
  for (int i = threadIdx.x; i < C_ * K2_; i += 256) lp[i] = pooled[b * C_ * K2_ + i];
  __syncthreads();

  float acc[K2_];
#pragma unroll
  for (int k = 0; k < K2_; k++) acc[k] = b_kp[o];
  const float* wr = w_kp + (size_t)o * C_;
  for (int c = 0; c < C_; c++) {
    const float wv = wr[c];
    const float* pc = &lp[c * K2_];
#pragma unroll
    for (int k = 0; k < K2_; k++) acc[k] += wv * pc[k];
  }
#pragma unroll
  for (int k = 0; k < K2_; k++) {   // gelu (tanh approx, jax default)
    const float v = acc[k];
    const float t = 0.7978845608028654f * (v + 0.044715f * v * v * v);
    acc[k] = 0.5f * v * (1.0f + tanhf(t));
  }
  float kj[K2_];
  float mean = 0.f;
#pragma unroll
  for (int j = 0; j < K2_; j++) {
    float s = b_kg[j];
#pragma unroll
    for (int k = 0; k < K2_; k++) s += w_kg[j * K2_ + k] * acc[k];
    kj[j] = s;
    mean += s;
  }
  mean *= (1.0f / K2_);
  const float factor = 1.0f / (1.0f + expf(-diff_ctrl[o]));
  float* out = ker + ((size_t)b * C_ + o) * K2_;
#pragma unroll
  for (int j = 0; j < K2_; j++) out[j] = kj[j] - factor * mean;
}

// ---------------------------------------------------------------------------
// 3) Convert both 256x256 projection weights to bf16 (WMMA A operands).
// ---------------------------------------------------------------------------
__global__ void cvt_weights(const float* __restrict__ wx, const float* __restrict__ wp,
                            bf16* __restrict__ wxb, bf16* __restrict__ wpb) {
  const int i = blockIdx.x * 256 + threadIdx.x;
  if (i < C_ * C_) {
    wxb[i] = (bf16)wx[i];
    wpb[i] = (bf16)wp[i];
  }
}

// ---------------------------------------------------------------------------
// 4) Transpose + convert x: [b][c][hw] f32 -> [b][hw][c] bf16 (WMMA B layout:
//    lane==column(hw), contiguous K(c) per lane). LDS 32x32 tile transpose.
// ---------------------------------------------------------------------------
__global__ void transpose_cvt(const float* __restrict__ x, bf16* __restrict__ xT) {
  // grid (HW_/32, C_/32, B_), block (32, 8)
  __shared__ bf16 tile[32][33];
  const int b = blockIdx.z;
  const int s0 = blockIdx.x * 32, c0 = blockIdx.y * 32;
  const int tx = threadIdx.x, ty = threadIdx.y;
#pragma unroll
  for (int it = 0; it < 4; ++it) {
    const int cc = c0 + ty + 8 * it;
    tile[ty + 8 * it][tx] = (bf16)x[((size_t)(b * C_ + cc)) * HW_ + s0 + tx];
  }
  __syncthreads();
#pragma unroll
  for (int it = 0; it < 4; ++it) {
    const int s = s0 + ty + 8 * it;
    xT[((size_t)b * HW_ + s) * C_ + c0 + tx] = tile[tx][ty + 8 * it];
  }
}

// ---------------------------------------------------------------------------
// 5) WMMA GEMM with async-to-LDS double buffering.
//    out[o][n] = sum_c A[o][c] * Bm[n][c] + bias[o]
//    Block tile M=64 x N=128, K-step 32, 8 waves (wave tile 16M x 64N).
//    Per K-step the block stages A(64x32 bf16) + B(128x32 bf16) in LDS via
//    GLOBAL_LOAD_ASYNC_TO_LDS_B128 (ASYNCcnt), double buffered so the next
//    tile's DMA overlaps this tile's WMMAs. LDS rows padded to 80B so the
//    16-lane fragment reads hit distinct banks (stride 20 banks, 16B aligned).
// ---------------------------------------------------------------------------
#define KSTEP    32
#define ROW_B    80                         // padded LDS row stride (bytes)
#define A_STAGE  (64 * ROW_B)               // 5120 B
#define B_STAGE  (128 * ROW_B)              // 10240 B
#define STAGE_B  (A_STAGE + B_STAGE)        // 15360 B

template <int STORE_BF16_T>
__global__ void gemm_wmma(const bf16* __restrict__ A,
                          const bf16* __restrict__ Bm,
                          const float* __restrict__ bias,
                          bf16* __restrict__ outT,
                          float* __restrict__ outF) {
  __shared__ __align__(16) char smem[2 * STAGE_B];

  const int b    = blockIdx.z;
  const int tid  = threadIdx.x;
  const int lane = tid & 31;
  const int wave = tid >> 5;
  const int mg   = wave & 3;    // 4 M-groups of 16 rows
  const int ng   = wave >> 2;   // 2 N-groups of 64 cols
  const int row0 = blockIdx.y * 64;          // block M origin
  const int n0   = blockIdx.x * 128;         // block N origin
  const int l16  = lane & 15;
  const int hi   = lane >> 4;

  const bf16* Bbase = Bm + (size_t)b * HW_ * C_;

  // --- per-thread async-copy chunk assignment (16B chunks) ---
  // A tile: 64 rows x 64B = 256 chunks -> 1 per thread
  const int a_m = tid >> 2, a_c = tid & 3;
  const char* a_src0 = (const char*)(A + (size_t)(row0 + a_m) * C_) + a_c * 16;
  const unsigned a_dst = (unsigned)(a_m * ROW_B + a_c * 16);
  // B tile: 128 rows x 64B = 512 chunks -> 2 per thread
  const int b0_n = tid >> 2,        b0_c = tid & 3;
  const int b1_n = 64 + (tid >> 2), b1_c = tid & 3;
  const char* b0_src0 = (const char*)(Bbase + (size_t)(n0 + b0_n) * C_) + b0_c * 16;
  const char* b1_src0 = (const char*)(Bbase + (size_t)(n0 + b1_n) * C_) + b1_c * 16;
  const unsigned b0_dst = (unsigned)(A_STAGE + b0_n * ROW_B + b0_c * 16);
  const unsigned b1_dst = (unsigned)(A_STAGE + b1_n * ROW_B + b1_c * 16);

  // LDS byte offset of smem (low 32 bits of the flat shared address)
  const unsigned smem_lds = (unsigned)(size_t)(&smem[0]);

  // fragment read pointers (generic->LDS, lowered to ds_load_b128)
  const int afrag_row  = mg * 16 + l16;
  char* const a_rd = smem + afrag_row * ROW_B + hi * 16;
  char* const b_rd = smem + A_STAGE + (ng * 64 + l16) * ROW_B + hi * 16;

  v8f acc0 = {}, acc1 = {}, acc2 = {}, acc3 = {};

  // prologue: stage 0 <- K block 0
  {
    const unsigned sb = smem_lds;
    asm volatile("global_load_async_to_lds_b128 %0, %1, off"
                 :: "v"(sb + a_dst), "v"((unsigned long long)a_src0) : "memory");
    asm volatile("global_load_async_to_lds_b128 %0, %1, off"
                 :: "v"(sb + b0_dst), "v"((unsigned long long)b0_src0) : "memory");
    asm volatile("global_load_async_to_lds_b128 %0, %1, off"
                 :: "v"(sb + b1_dst), "v"((unsigned long long)b1_src0) : "memory");
  }

  const int NSTEPS = C_ / KSTEP;  // 8
  for (int i = 0; i < NSTEPS; ++i) {
    if (i + 1 < NSTEPS) {
      // issue next stage (K advances by 32 halves = 64 bytes)
      const unsigned sb = smem_lds + ((i + 1) & 1) * STAGE_B;
      const size_t koff = (size_t)(i + 1) * 64;
      asm volatile("global_load_async_to_lds_b128 %0, %1, off"
                   :: "v"(sb + a_dst), "v"((unsigned long long)(a_src0 + koff)) : "memory");
      asm volatile("global_load_async_to_lds_b128 %0, %1, off"
                   :: "v"(sb + b0_dst), "v"((unsigned long long)(b0_src0 + koff)) : "memory");
      asm volatile("global_load_async_to_lds_b128 %0, %1, off"
                   :: "v"(sb + b1_dst), "v"((unsigned long long)(b1_src0 + koff)) : "memory");
      // 3 newer asyncs may stay in flight; ASYNCcnt is in-order for loads,
      // so <=3 means this wave's current-stage chunks have landed.
      asm volatile("s_wait_asynccnt 0x3" ::: "memory");
    } else {
      asm volatile("s_wait_asynccnt 0x0" ::: "memory");
    }
    __syncthreads();   // all waves' chunks for stage i are now visible

    {
      const int st = (i & 1) * STAGE_B;
      // A fragment: lanes<16 K halves {0..7,16..23}, lanes>=16 {8..15,24..31}
      const v8bf alo = *(const v8bf*)(a_rd + st);
      const v8bf ahi = *(const v8bf*)(a_rd + st + 32);
      v16bf af;
#pragma unroll
      for (int e = 0; e < 8; e++) { af[e] = alo[e]; af[e + 8] = ahi[e]; }

      // B fragments: lane = column, K halves {0..7,16..23} / {8..15,24..31} (same as A)
      v16bf bf[4];
#pragma unroll
      for (int t = 0; t < 4; t++) {
        const char* pb = b_rd + st + t * (16 * ROW_B);
        const v8bf blo = *(const v8bf*)(pb);
        const v8bf bhi = *(const v8bf*)(pb + 32);
#pragma unroll
        for (int e = 0; e < 8; e++) { bf[t][e] = blo[e]; bf[t][e + 8] = bhi[e]; }
      }
      acc0 = __builtin_amdgcn_wmma_f32_16x16x32_bf16(false, af, false, bf[0], (short)0, acc0, false, false);
      acc1 = __builtin_amdgcn_wmma_f32_16x16x32_bf16(false, af, false, bf[1], (short)0, acc1, false, false);
      acc2 = __builtin_amdgcn_wmma_f32_16x16x32_bf16(false, af, false, bf[2], (short)0, acc2, false, false);
      acc3 = __builtin_amdgcn_wmma_f32_16x16x32_bf16(false, af, false, bf[3], (short)0, acc3, false, false);
    }
    __syncthreads();   // protect stage buffer before the next overwrite
  }

  // D layout: lane column n = n0 + ng*64 + t*16 + l16, VGPR v -> row v + 8*hi.
  const int rowA = row0 + mg * 16;
  float bv[8];
#pragma unroll
  for (int v = 0; v < 8; v++) bv[v] = bias[rowA + hi * 8 + v];

#pragma unroll
  for (int t = 0; t < 4; t++) {
    const v8f acc = (t == 0) ? acc0 : (t == 1) ? acc1 : (t == 2) ? acc2 : acc3;
    const int n = n0 + ng * 64 + t * 16 + l16;
    if (STORE_BF16_T) {
      v8bf pack;
#pragma unroll
      for (int v = 0; v < 8; v++) pack[v] = (bf16)(acc[v] + bv[v]);
      // 8 consecutive output channels -> single 16B store in [hw][c] layout
      *(v8bf*)(outT + ((size_t)b * HW_ + n) * C_ + rowA + hi * 8) = pack;
    } else {
#pragma unroll
      for (int v = 0; v < 8; v++)
        outF[((size_t)(b * C_ + rowA + hi * 8 + v)) * HW_ + n] = acc[v] + bv[v];
    }
  }
}

// ---------------------------------------------------------------------------
// 6) Per-sample dynamic depthwise 3x3 conv on transposed bf16 activations.
//    [hw][c] layout => all 9 taps and the store are coalesced along c.
// ---------------------------------------------------------------------------
__global__ void dwconv_kernel(const bf16* __restrict__ xpT, const float* __restrict__ ker,
                              bf16* __restrict__ yT) {
  // grid (HW_, B_), block 256 (= C_)
  const int hw = blockIdx.x;
  const int b  = blockIdx.y;
  const int c  = threadIdx.x;
  const int h = hw / W_, w = hw % W_;
  const float* kc = ker + ((size_t)b * C_ + c) * K2_;
  const bf16* base = xpT + (size_t)b * HW_ * C_;
  float s = 0.f;
#pragma unroll
  for (int k1 = 0; k1 < 3; k1++) {
    const int hh = h + k1 - 1;
    if (hh < 0 || hh >= H_) continue;
#pragma unroll
    for (int k2 = 0; k2 < 3; k2++) {
      const int ww = w + k2 - 1;
      if (ww < 0 || ww >= W_) continue;
      s += kc[k1 * 3 + k2] * (float)base[(size_t)(hh * W_ + ww) * C_ + c];
    }
  }
  yT[((size_t)b * HW_ + hw) * C_ + c] = (bf16)s;
}

// ---------------------------------------------------------------------------
// Host launch
// ---------------------------------------------------------------------------
extern "C" void kernel_launch(void* const* d_in, const int* in_sizes, int n_in,
                              void* d_out, int out_size, void* d_ws, size_t ws_size,
                              hipStream_t stream) {
  const float* x         = (const float*)d_in[0];
  const float* w_xproj   = (const float*)d_in[1];
  const float* b_xproj   = (const float*)d_in[2];
  const float* w_proj    = (const float*)d_in[3];
  const float* b_proj    = (const float*)d_in[4];
  const float* w_kp      = (const float*)d_in[5];
  const float* b_kp      = (const float*)d_in[6];
  const float* w_kg      = (const float*)d_in[7];
  const float* b_kg      = (const float*)d_in[8];
  const float* diff_ctrl = (const float*)d_in[9];

  char* ws = (char*)d_ws;
  // workspace layout (bytes); yT reuses the xT buffer (xT dead after GEMM1)
  bf16*  wAb    = (bf16*)(ws + 0);                 //   131072
  bf16*  wPb    = (bf16*)(ws + 131072);            //   131072
  float* pooled = (float*)(ws + 262144);           //   147456
  float* ker    = (float*)(ws + 409600);           //   147456
  bf16*  xT     = (bf16*)(ws + 557056);            // 75497472
  bf16*  xpT    = (bf16*)(ws + 76054528);          // 75497472 -> total ~151.5 MB
  bf16*  yT     = xT;

  float* out = (float*)d_out;

  pool_kernel<<<dim3(K2_, C_, B_), 256, 0, stream>>>(x, pooled);
  ker_synth<<<B_, 256, 0, stream>>>(pooled, w_kp, b_kp, w_kg, b_kg, diff_ctrl, ker);
  cvt_weights<<<(C_ * C_ + 255) / 256, 256, 0, stream>>>(w_xproj, w_proj, wAb, wPb);
  transpose_cvt<<<dim3(HW_ / 32, C_ / 32, B_), dim3(32, 8), 0, stream>>>(x, xT);
  gemm_wmma<1><<<dim3(HW_ / 128, C_ / 64, B_), 256, 0, stream>>>(wAb, xT, b_xproj, xpT, nullptr);
  dwconv_kernel<<<dim3(HW_, B_), 256, 0, stream>>>(xpT, ker, yT);
  gemm_wmma<0><<<dim3(HW_ / 128, C_ / 64, B_), 256, 0, stream>>>(wPb, yT, b_proj, nullptr, out);
}
